// NLBlockND_multicross_block_81114752352479
// MI455X (gfx1250) — matrix operand order repaired and checked by reading, not compiled, not run on hardware
//
#include <hip/hip_runtime.h>
#include <hip/hip_bf16.h>

// ---------------------------------------------------------------------------
// Non-local block (flash-attention style) for gfx1250 / MI455X.
// bf16 WMMA (v_wmma_f32_16x16x32_bf16) with f32 accumulation; f32 softmax/BN.
// CDNA5 paths: v_wmma_*, global_load_async_to_lds_b64 (+ s_wait_asynccnt),
// global_prefetch_b8, b128 LDS fragment loads.
// ---------------------------------------------------------------------------

typedef __attribute__((ext_vector_type(16))) __bf16    v16bf;
typedef __attribute__((ext_vector_type(8)))  float     v8f;
typedef __attribute__((ext_vector_type(8)))  unsigned  v8u;
typedef __attribute__((ext_vector_type(4)))  unsigned  v4u;

#define DEVINL __device__ __forceinline__

DEVINL unsigned short f2bf(float f) {
  __bf16 h = (__bf16)f;                       // native cvt on gfx1250
  return __builtin_bit_cast(unsigned short, h);
}
DEVINL unsigned pack2(float lo, float hi) {
  return (unsigned)f2bf(lo) | ((unsigned)f2bf(hi) << 16);
}
DEVINL v8f vzero() {
  v8f z;
#pragma unroll
  for (int i = 0; i < 8; ++i) z[i] = 0.f;
  return z;
}
DEVINL v8f wmma_bf16(v8u a, v8u b, v8f c) {
  return __builtin_amdgcn_wmma_f32_16x16x32_bf16(
      false, __builtin_bit_cast(v16bf, a),
      false, __builtin_bit_cast(v16bf, b),
      (short)0, c, false, false);
}
// B-fragment: two 8-bf16 runs at +0 and +16 elements: two b128 loads.
DEVINL v8u ld_bfrag(const unsigned short* p) {
  v4u a = *(const v4u*)p;
  v4u b = *(const v4u*)(p + 16);
  v8u r;
  r[0] = a[0]; r[1] = a[1]; r[2] = a[2]; r[3] = a[3];
  r[4] = b[0]; r[5] = b[1]; r[6] = b[2]; r[7] = b[3];
  return r;
}
// A-fragment: two 8-bf16 runs at +0 and +16 elements: two b128 loads.
DEVINL v8u ld_afrag(const unsigned short* p) {
  v4u a = *(const v4u*)p;
  v4u b = *(const v4u*)(p + 16);
  v8u r;
  r[0] = a[0]; r[1] = a[1]; r[2] = a[2]; r[3] = a[3];
  r[4] = b[0]; r[5] = b[1]; r[6] = b[2]; r[7] = b[3];
  return r;
}
// Async global->LDS copy (ASYNCcnt-tracked), 8 bytes per lane.
DEVINL void async_load_b64(const void* gptr, void* lptr) {
  unsigned lds = (unsigned)(unsigned long long)lptr;  // LDS offset = addr[31:0]
  asm volatile("global_load_async_to_lds_b64 %0, %1, off"
               :: "v"(lds), "v"(gptr) : "memory");
}
DEVINL void wait_asynccnt0() {
#if __has_builtin(__builtin_amdgcn_s_wait_asynccnt)
  __builtin_amdgcn_s_wait_asynccnt(0);
#else
  asm volatile("s_wait_asynccnt 0x0" ::: "memory");
#endif
}

// ---------------------------------------------------------------------------
// Kernel 1: 1x1 conv projection.  out[o,n] = sum_c w[o,c] x[c,n] + b[o]
// layoutNC==0 -> out [B][C][N] (theta, g);  ==1 -> [B][N][C] (phi^T).
// ---------------------------------------------------------------------------
__global__ __launch_bounds__(256)
void proj_kernel(const float* __restrict__ x, const float* __restrict__ w,
                 const float* __restrict__ bias, unsigned short* __restrict__ out,
                 int layoutNC) {
  constexpr int CIN = 256, CINT = 128, N = 4096;
  constexpr int CS = 72;                     // 16B-aligned rows, benign 2-way
  __shared__ unsigned short sX[128 * CS];    // x tile, [n][c] transposed, bf16

  const int b  = blockIdx.y;
  const int n0 = blockIdx.x * 128;
  const int t  = threadIdx.x;
  const int wv = t >> 5, ln = t & 31;
  const int m  = ln & 15, hi = ln >> 4;

  v8f acc[8];
#pragma unroll
  for (int s = 0; s < 8; ++s) acc[s] = vzero();

  const float* wrow = &w[(size_t)(wv * 16 + m) * CIN];

  for (int cc = 0; cc < CIN; cc += 64) {
    __syncthreads();
    // x[b][cc..cc+63][n0..n0+127] -> sX[n][c], float4 global loads
#pragma unroll
    for (int it = 0; it < 8; ++it) {
      int e = (t + it * 256) * 4;            // 8192 elements
      int r = e >> 7, col = e & 127;         // r: c offset, col: n offset (x4)
      float4 xv = *(const float4*)&x[((size_t)b * CIN + cc + r) * N + n0 + col];
      sX[(col + 0) * CS + r] = f2bf(xv.x);
      sX[(col + 1) * CS + r] = f2bf(xv.y);
      sX[(col + 2) * CS + r] = f2bf(xv.z);
      sX[(col + 3) * CS + r] = f2bf(xv.w);
    }
    __syncthreads();
#pragma unroll
    for (int ks = 0; ks < 2; ++ks) {
      // A fragment from W (f32 -> packed bf16), two 32B runs
      int kb = cc + ks * 32 + hi * 8;
      float4 w0 = *(const float4*)(wrow + kb);
      float4 w1 = *(const float4*)(wrow + kb + 4);
      float4 w2 = *(const float4*)(wrow + kb + 16);
      float4 w3 = *(const float4*)(wrow + kb + 20);
      v8u af;
      af[0] = pack2(w0.x, w0.y); af[1] = pack2(w0.z, w0.w);
      af[2] = pack2(w1.x, w1.y); af[3] = pack2(w1.z, w1.w);
      af[4] = pack2(w2.x, w2.y); af[5] = pack2(w2.z, w2.w);
      af[6] = pack2(w3.x, w3.y); af[7] = pack2(w3.z, w3.w);
#pragma unroll
      for (int sub = 0; sub < 8; ++sub) {
        v8u bfr = ld_bfrag(&sX[(sub * 16 + m) * CS + ks * 32 + hi * 8]);
        acc[sub] = wmma_bf16(af, bfr, acc[sub]);
      }
    }
  }
#pragma unroll
  for (int sub = 0; sub < 8; ++sub) {
#pragma unroll
    for (int r = 0; r < 8; ++r) {
      int o = wv * 16 + r + hi * 8;
      int n = n0 + sub * 16 + m;
      float vo = acc[sub][r] + bias[o];
      size_t idx = layoutNC ? ((size_t)b * N + n) * CINT + o
                            : ((size_t)b * CINT + o) * N + n;
      out[idx] = f2bf(vo);
    }
  }
}

// ---------------------------------------------------------------------------
// Kernel 2: flash attention.
// ---------------------------------------------------------------------------
__global__ __launch_bounds__(256)
void attn_kernel(const unsigned short* __restrict__ thetaCN,
                 const unsigned short* __restrict__ phiNC,
                 const unsigned short* __restrict__ gCN,
                 unsigned short* __restrict__ yNC) {
  constexpr int C = 128, N = 4096, JT = 64;
  constexpr int TS = 136;                    // sK stride [j][c]
  constexpr int GS = 72;                     // sV stride [c][j]
  constexpr int PS = 72;                     // P scratch stride [i][j]
  __shared__ unsigned short sK[JT * TS];
  __shared__ unsigned short sV[C * GS];
  __shared__ unsigned short sP[8 * 16 * PS];

  const int b  = blockIdx.y;
  const int i0 = blockIdx.x * 128;
  const int t  = threadIdx.x;
  const int wv = t >> 5, ln = t & 31;
  const int m  = ln & 15, hi = ln >> 4;

  // Q fragments, resident for the whole kernel.
  const unsigned short* qrow = &phiNC[((size_t)b * N + i0 + wv * 16 + m) * C];
  v8u qf[4];
#pragma unroll
  for (int kk = 0; kk < 4; ++kk) qf[kk] = ld_afrag(qrow + kk * 32 + hi * 8);

  v8f yacc[8];
#pragma unroll
  for (int s = 0; s < 8; ++s) yacc[s] = vzero();
  float mrow[8], lrow[8];
#pragma unroll
  for (int r = 0; r < 8; ++r) { mrow[r] = -3.0e38f; lrow[r] = 0.f; }

  unsigned short* myP = &sP[wv * 16 * PS];

  for (int j0 = 0; j0 < N; j0 += JT) {
    __syncthreads();
    if (j0 + JT < N) {                       // global_prefetch_b8 next tiles
      __builtin_prefetch(&thetaCN[((size_t)b * C + (t & 127)) * N + j0 + JT], 0, 0);
      __builtin_prefetch(&gCN[((size_t)b * C + (t & 127)) * N + j0 + JT], 0, 0);
    }
#pragma unroll
    for (int it = 0; it < 8; ++it) {         // 128c x 64j, 4 j per thread
      int e = (t + it * 256) * 4;
      int c = e >> 6, j = e & 63;
      size_t src = ((size_t)b * C + c) * N + j0 + j;
      // V tile: straight copy -> async DMA into LDS (no VGPR round-trip)
      async_load_b64(&gCN[src], &sV[c * GS + j]);
      // K tile: transpose c<->j through VGPRs
      uint2 tv = *(const uint2*)&thetaCN[src];
      sK[(j + 0) * TS + c] = (unsigned short)(tv.x);
      sK[(j + 1) * TS + c] = (unsigned short)(tv.x >> 16);
      sK[(j + 2) * TS + c] = (unsigned short)(tv.y);
      sK[(j + 3) * TS + c] = (unsigned short)(tv.y >> 16);
    }
    wait_asynccnt0();
    __syncthreads();

    // S = Q * K^T (4 j-subtiles of 16, K=128 over c)
    v8f s[4];
#pragma unroll
    for (int sub = 0; sub < 4; ++sub) {
      s[sub] = vzero();
#pragma unroll
      for (int kk = 0; kk < 4; ++kk) {
        v8u bfr = ld_bfrag(&sK[(sub * 16 + m) * TS + kk * 32 + hi * 8]);
        s[sub] = wmma_bf16(qf[kk], bfr, s[sub]);
      }
    }

    // online softmax (row r spans 16 lanes of a half-group)
#pragma unroll
    for (int r = 0; r < 8; ++r) {
      float mx = fmaxf(fmaxf(s[0][r], s[1][r]), fmaxf(s[2][r], s[3][r]));
#pragma unroll
      for (int d = 8; d >= 1; d >>= 1) mx = fmaxf(mx, __shfl_xor(mx, d, 32));
      float mnew  = fmaxf(mrow[r], mx);
      float scale = __expf(mrow[r] - mnew);
      float psum  = 0.f;
#pragma unroll
      for (int sub = 0; sub < 4; ++sub) {
        float p = __expf(s[sub][r] - mnew);
        s[sub][r] = p;
        psum += p;
      }
#pragma unroll
      for (int d = 8; d >= 1; d >>= 1) psum += __shfl_xor(psum, d, 32);
      lrow[r] = lrow[r] * scale + psum;
      mrow[r] = mnew;
#pragma unroll
      for (int sub = 0; sub < 8; ++sub) yacc[sub][r] *= scale;
    }

    // transpose P (C-layout -> A-layout) through wave-private LDS
#pragma unroll
    for (int sub = 0; sub < 4; ++sub)
#pragma unroll
      for (int r = 0; r < 8; ++r)
        myP[(r + hi * 8) * PS + sub * 16 + m] = f2bf(s[sub][r]);

    v8u pf[2];
#pragma unroll
    for (int kk = 0; kk < 2; ++kk)
      pf[kk] = ld_afrag(&myP[m * PS + kk * 32 + hi * 8]);

    // Y += P * G^T (8 c-subtiles, K=64 over j)
#pragma unroll
    for (int sub = 0; sub < 8; ++sub)
#pragma unroll
      for (int kk = 0; kk < 2; ++kk) {
        v8u bfr = ld_bfrag(&sV[(sub * 16 + m) * GS + kk * 32 + hi * 8]);
        yacc[sub] = wmma_bf16(pf[kk], bfr, yacc[sub]);
      }
  }

#pragma unroll
  for (int r = 0; r < 8; ++r) {
    float inv = 1.f / lrow[r];
    int i = i0 + wv * 16 + r + hi * 8;
#pragma unroll
    for (int sub = 0; sub < 8; ++sub) {
      int c = sub * 16 + m;
      yNC[((size_t)b * N + i) * C + c] = f2bf(yacc[sub][r] * inv);
    }
  }
}

// ---------------------------------------------------------------------------
// Kernel 3: z^T[i,o] = sum_c y^T[i,c] w_z[o,c] + b_z[o] (WMMA, K=128)
// + deterministic per-block BN partial sums.
// ---------------------------------------------------------------------------
__global__ __launch_bounds__(256)
void zgemm_kernel(const unsigned short* __restrict__ yNC,
                  const float* __restrict__ wz, const float* __restrict__ bz,
                  float* __restrict__ zNC, float* __restrict__ part) {
  constexpr int C = 128, N = 4096;
  constexpr int WSS = 136;
  __shared__ unsigned short sW[C * WSS];     // w_z as B-matrix [o][c], bf16
  __shared__ float sSum[128], sSsq[128];

  const int b  = blockIdx.y;
  const int i0 = blockIdx.x * 128;
  const int t  = threadIdx.x;
  const int wv = t >> 5, ln = t & 31;
  const int m  = ln & 15, hi = ln >> 4;

#pragma unroll
  for (int it = 0; it < 16; ++it) {          // 128x128 f32 -> bf16 pairs
    int e = (t + it * 256) * 4;
    int o = e >> 7, c = e & 127;
    float4 wv4 = *(const float4*)&wz[(size_t)o * C + c];
    *(uint2*)&sW[o * WSS + c] = make_uint2(pack2(wv4.x, wv4.y), pack2(wv4.z, wv4.w));
  }
  if (t < 128) { sSum[t] = 0.f; sSsq[t] = 0.f; }
  __syncthreads();

  const unsigned short* yrow = &yNC[((size_t)b * N + i0 + wv * 16 + m) * C];
  v8u af[4];
#pragma unroll
  for (int kk = 0; kk < 4; ++kk) af[kk] = ld_afrag(yrow + kk * 32 + hi * 8);

  v8f acc[8];
#pragma unroll
  for (int s = 0; s < 8; ++s) acc[s] = vzero();
#pragma unroll
  for (int sub = 0; sub < 8; ++sub)
#pragma unroll
    for (int kk = 0; kk < 4; ++kk) {
      v8u bfr = ld_bfrag(&sW[(sub * 16 + m) * WSS + kk * 32 + hi * 8]);
      acc[sub] = wmma_bf16(af[kk], bfr, acc[sub]);
    }

#pragma unroll
  for (int sub = 0; sub < 8; ++sub) {
    int o = sub * 16 + m;
    float bb = bz[o], ssum = 0.f, ssq = 0.f;
#pragma unroll
    for (int r = 0; r < 8; ++r) {
      int i = i0 + wv * 16 + r + hi * 8;
      float z = acc[sub][r] + bb;
      zNC[((size_t)b * N + i) * C + o] = z;
      ssum += z;
      ssq  += z * z;
    }
    atomicAdd(&sSum[o], ssum);               // block-local ds_add_f32
    atomicAdd(&sSsq[o], ssq);
  }
  __syncthreads();
  if (t < 128) {
    size_t blk = (size_t)(b * 32 + blockIdx.x) * 256;
    part[blk + t]       = sSum[t];
    part[blk + 128 + t] = sSsq[t];
  }
}

// Kernel 4: deterministic fixed-order reduction of 128 block partials.
__global__ __launch_bounds__(256)
void redstats_kernel(const float* __restrict__ part, float* __restrict__ stats) {
  int t = threadIdx.x;
  float s = 0.f;
  for (int k = 0; k < 128; ++k) s += part[(size_t)k * 256 + t];
  stats[t] = s;
}

// Kernel 5: BN normalize + transpose [B][N][C] -> [B][C][N] via LDS tile.
__global__ __launch_bounds__(256)
void bn_kernel(const float* __restrict__ zNC, const float* __restrict__ stats,
               const float* __restrict__ gamma, const float* __restrict__ beta,
               float* __restrict__ out) {
  constexpr int C = 128, N = 4096;
  __shared__ float tile[32][33];
  const int b  = blockIdx.z;
  const int o0 = blockIdx.y * 32;
  const int i0 = blockIdx.x * 32;
  const int tx = threadIdx.x, ty = threadIdx.y;  // 32 x 8
#pragma unroll
  for (int s = 0; s < 4; ++s) {
    int i = i0 + ty + s * 8;
    tile[ty + s * 8][tx] = zNC[((size_t)b * N + i) * C + o0 + tx];
  }
  __syncthreads();
  const float invcnt = 1.f / (4.f * 4096.f);
#pragma unroll
  for (int s = 0; s < 4; ++s) {
    int o = o0 + ty + s * 8;
    float mean = stats[o] * invcnt;
    float var  = stats[128 + o] * invcnt - mean * mean;
    float sc   = gamma[o] * rsqrtf(var + 1e-5f);
    float val  = tile[tx][ty + s * 8];
    out[((size_t)b * C + o) * N + i0 + tx] = (val - mean) * sc + beta[o];
  }
}

// ---------------------------------------------------------------------------
extern "C" void kernel_launch(void* const* d_in, const int* in_sizes, int n_in,
                              void* d_out, int out_size, void* d_ws, size_t ws_size,
                              hipStream_t stream) {
  const float* x_this  = (const float*)d_in[0];
  const float* x_other = (const float*)d_in[1];
  const float* w_g     = (const float*)d_in[2];
  const float* b_g     = (const float*)d_in[3];
  const float* w_th    = (const float*)d_in[4];
  const float* b_th    = (const float*)d_in[5];
  const float* w_ph    = (const float*)d_in[6];
  const float* b_ph    = (const float*)d_in[7];
  const float* w_z     = (const float*)d_in[8];
  const float* b_z     = (const float*)d_in[9];
  const float* gamma   = (const float*)d_in[10];
  const float* beta    = (const float*)d_in[11];
  float* out = (float*)d_out;

  char* ws = (char*)d_ws;
  const size_t MB = 1024u * 1024u;
  unsigned short* thetaCN = (unsigned short*)(ws + 0);        // 4 MB bf16 [B][C][N]
  unsigned short* phiNC   = (unsigned short*)(ws + 4 * MB);   // 4 MB bf16 [B][N][C]
  unsigned short* gCN     = (unsigned short*)(ws + 8 * MB);   // 4 MB bf16 [B][C][N]
  unsigned short* yNC     = (unsigned short*)(ws + 12 * MB);  // 4 MB bf16 [B][N][C]
  float*          zNC     = (float*)(ws + 16 * MB);           // 8 MB f32 [B][N][C]
  float*          part    = (float*)(ws + 24 * MB);           // 128 KB partials
  float*          stats   = (float*)(ws + 24 * MB + 128u * 1024u); // 1 KB

  dim3 gProj(32, 4), blk(256);
  proj_kernel<<<gProj, blk, 0, stream>>>(x_this,  w_th, b_th, thetaCN, 0);
  proj_kernel<<<gProj, blk, 0, stream>>>(x_other, w_ph, b_ph, phiNC,   1);
  proj_kernel<<<gProj, blk, 0, stream>>>(x_this,  w_g,  b_g,  gCN,     0);

  attn_kernel<<<dim3(32, 4), blk, 0, stream>>>(thetaCN, phiNC, gCN, yNC);
  zgemm_kernel<<<dim3(32, 4), blk, 0, stream>>>(yNC, w_z, b_z, zNC, part);
  redstats_kernel<<<1, 256, 0, stream>>>(part, stats);
  bn_kernel<<<dim3(128, 4, 4), dim3(32, 8), 0, stream>>>(zNC, stats, gamma, beta, out);
}
